// VFPG_theirs_87935160419019
// MI455X (gfx1250) — matrix operand order repaired and checked by reading, not compiled, not run on hardware
//
#include <hip/hip_runtime.h>
#include <stdint.h>

// Problem constants (from reference)
#define M_SEQ   2048
#define T_STEPS 64
#define IN_F    16
#define H_DIM   256
#define NCOMP   64
#define OUT_DIM 192
#define ROWS    (M_SEQ * T_STEPS)   // 131072

typedef __attribute__((ext_vector_type(16))) __bf16 v16bf;
typedef __attribute__((ext_vector_type(8)))  float  v8f;

union BfPack { uint4 q[2]; v16bf v; };

__device__ __forceinline__ __bf16 f2bf(float f) {
  unsigned u = __builtin_bit_cast(unsigned, f);
  unsigned r = u + 0x7FFFu + ((u >> 16) & 1u);   // round-to-nearest-even
  unsigned short h = (unsigned short)(r >> 16);
  return __builtin_bit_cast(__bf16, h);
}

__device__ __forceinline__ v8f wmma_bf16(v16bf a, v16bf b, v8f c) {
  // D = A(16x32 bf16) * B(32x16 bf16) + C(16x16 f32)
  return __builtin_amdgcn_wmma_f32_16x16x32_bf16(
      /*neg_a=*/false, a, /*neg_b=*/false, b,
      /*c_mod=*/(short)0, c, /*reuse_a=*/false, /*reuse_b=*/false);
}

__device__ __forceinline__ float sigm(float x) { return 1.0f / (1.0f + __expf(-x)); }

__device__ __forceinline__ unsigned hash_u32(unsigned x) {
  x ^= x >> 16; x *= 0x7feb352dU;
  x ^= x >> 15; x *= 0x846ca68bU;
  x ^= x >> 16; return x;
}

// ---------------------------------------------------------------------------
// Fully-unrolled, software-pipelined GEMM tile block.
//   A: 16xK bf16 tile in LDS, row-major [16][aRowB bytes]
//      (lane = M, lane half selects contiguous K 0-15 / 16-31, same order as B)
//   B: W row-major [n][ldk] bf16 in global; B = W^T slices
//      (lane = N, lane half selects contiguous K 0-15 / 16-31 => 2x16B loads)
//   C: CT accumulator tiles, columns nb + c*16.
// One-stage lookahead on both A and B so the next tile's loads are issued
// before the current WMMA -> backend can use partial s_wait_loadcnt.
// ---------------------------------------------------------------------------
template <int KT, int CT>
__device__ __forceinline__ void gemm_tiles(const char* aBase, int aRowB,
                                           const __bf16* W, int ldk,
                                           int nb, int lane, v8f* acc) {
  const int r15  = lane & 15;
  const int half = (lane >> 4) & 1;
  const char* aRow = aBase + r15 * aRowB + half * 32;
  const char* bp[CT];
#pragma unroll
  for (int c = 0; c < CT; ++c)
    bp[c] = (const char*)W + ((size_t)(nb + c * 16 + r15) * ldk + half * 16) * 2;

  BfPack a0, a1, b0, b1;
  a0.q[0] = *(const uint4*)(aRow + 0);
  a0.q[1] = *(const uint4*)(aRow + 16);
  b0.q[0] = *(const uint4*)(bp[0] + 0);
  b0.q[1] = *(const uint4*)(bp[0] + 16);

#pragma unroll
  for (int kt = 0; kt < KT; ++kt) {
    if (kt + 1 < KT) {   // prefetch next A tile from LDS
      a1.q[0] = *(const uint4*)(aRow + (kt + 1) * 64);
      a1.q[1] = *(const uint4*)(aRow + (kt + 1) * 64 + 16);
    }
#pragma unroll
    for (int c = 0; c < CT; ++c) {
      if (c + 1 < CT) {  // prefetch next B tile from global
        b1.q[0] = *(const uint4*)(bp[c + 1] + kt * 64);
        b1.q[1] = *(const uint4*)(bp[c + 1] + kt * 64 + 16);
      } else if (kt + 1 < KT) {
        b1.q[0] = *(const uint4*)(bp[0] + (kt + 1) * 64);
        b1.q[1] = *(const uint4*)(bp[0] + (kt + 1) * 64 + 16);
      }
      acc[c] = wmma_bf16(a0.v, b0.v, acc[c]);
      if (c + 1 < CT || kt + 1 < KT) b0 = b1;
    }
    if (kt + 1 < KT) a0 = a1;
  }
}

// ---------------------------------------------------------------------------
// Kernel 1: convert weights / inputs to bf16 (pad K=16 -> 32 for layer 0)
// region sizes: zpad 4194304 | Wih0pad 32768 | Whh0 262144 | Wih1 262144 |
//               Whh1 262144 | fcW 49152   => total 5062656 = 19776 * 256
// ---------------------------------------------------------------------------
__global__ void prep_kernel(const float* __restrict__ z,
                            const float* __restrict__ Wih0,
                            const float* __restrict__ Whh0,
                            const float* __restrict__ Wih1,
                            const float* __restrict__ Whh1,
                            const float* __restrict__ fcW,
                            __bf16* __restrict__ zpad,
                            __bf16* __restrict__ wih0p,
                            __bf16* __restrict__ whh0b,
                            __bf16* __restrict__ wih1b,
                            __bf16* __restrict__ whh1b,
                            __bf16* __restrict__ fcwb) {
  long i = (long)blockIdx.x * 256 + threadIdx.x;
  if (i < 4194304L) {                      // z [131072][16] -> [131072][32]
    long r = i >> 5; int k = (int)(i & 31);
    zpad[i] = (k < IN_F) ? f2bf(z[r * IN_F + k]) : f2bf(0.0f);
    return;
  }
  i -= 4194304L;
  if (i < 32768L) {                        // Wih0 [1024][16] -> [1024][32]
    long n = i >> 5; int k = (int)(i & 31);
    wih0p[i] = (k < IN_F) ? f2bf(Wih0[n * IN_F + k]) : f2bf(0.0f);
    return;
  }
  i -= 32768L;
  if (i < 262144L) { whh0b[i] = f2bf(Whh0[i]); return; }
  i -= 262144L;
  if (i < 262144L) { wih1b[i] = f2bf(Wih1[i]); return; }
  i -= 262144L;
  if (i < 262144L) { whh1b[i] = f2bf(Whh1[i]); return; }
  i -= 262144L;
  if (i < 49152L)  { fcwb[i] = f2bf(fcW[i]); }
}

// ---------------------------------------------------------------------------
// Kernel 2: one LSTM layer. Block = 16 sequences, 256 threads = 8 waves.
// Wave w owns gate columns [w*128, w*128+128) as 8 WMMA C-tiles.
// Per timestep: gates = x_t @ Wih^T + h @ Whh^T  (WMMA bf16, f32 acc),
// then fused cell update in fp32; c in registers, h in LDS bf16.
// Dynamic LDS: s_x 8KB | s_h 8KB | s_g 64KB  (80KB of the 320KB WGP pool)
// ---------------------------------------------------------------------------
__global__ void lstm_layer_kernel(const __bf16* __restrict__ X, int F,
                                  const __bf16* __restrict__ Wih,
                                  const __bf16* __restrict__ Whh,
                                  const float* __restrict__ bih,
                                  const float* __restrict__ bhh,
                                  __bf16* __restrict__ Hout) {
  extern __shared__ __align__(16) char smem[];
  char*  s_x = smem;                       // [16][F*2] bytes (<= 8KB)
  char*  s_h = smem + 8192;                // [16][512] bytes (bf16 h)
  float* s_g = (float*)(smem + 16384);     // [16][1024] gates f32

  const int tid  = threadIdx.x;
  const int lane = tid & 31;
  const int w    = tid >> 5;               // 0..7
  const int mBase = blockIdx.x * 16;
  const int xRowB = F * 2;                 // LDS x row bytes (64 or 512)

  // zero initial hidden state in LDS
  for (int i = tid; i < 16 * H_DIM; i += 256) ((__bf16*)s_h)[i] = f2bf(0.0f);

  float c_reg[16];
#pragma unroll
  for (int m = 0; m < 16; ++m) c_reg[m] = 0.0f;

  const int hc = tid;                      // hidden column owned in cell phase
  const float bi = bih[hc]           + bhh[hc];
  const float bf = bih[256 + hc]     + bhh[256 + hc];
  const float bg = bih[512 + hc]     + bhh[512 + hc];
  const float bo = bih[768 + hc]     + bhh[768 + hc];

  const v8f vzero = {0.f, 0.f, 0.f, 0.f, 0.f, 0.f, 0.f, 0.f};

  for (int t = 0; t < T_STEPS; ++t) {
    __syncthreads();   // guard s_x / s_g reuse across iterations

    // stage x_t tile [16][F] bf16 into LDS (16B chunks)
    const int nCh = (16 * F) >> 3;         // # of 8-bf16 chunks
    for (int i = tid; i < nCh; i += 256) {
      int m  = i / (F >> 3);
      int co = i % (F >> 3);
      *(uint4*)(s_x + m * xRowB + co * 16) =
          *(const uint4*)(X + (size_t)(mBase + m) * T_STEPS * F +
                          (size_t)t * F + co * 8);
    }
    // prefetch next step's x tile into cache (global_prefetch_b8)
    if (t + 1 < T_STEPS) {
      int m = tid & 15, chunk = tid >> 4;  // 16 chunks per row
      const __bf16* px = X + (size_t)(mBase + m) * T_STEPS * F +
                         (size_t)(t + 1) * F + (size_t)chunk * (F >> 4);
      __builtin_prefetch((const void*)px, 0, 3);
    }
    __syncthreads();   // x staged; h from previous cell phase visible

    v8f acc[8];
#pragma unroll
    for (int c = 0; c < 8; ++c) acc[c] = vzero;

    // Phase 1: x_t @ Wih^T   (K = 32 for layer 0, 256 for layer 1)
    if (F == 32) gemm_tiles<1, 8>(s_x, 64, Wih, 32, w * 128, lane, acc);
    else         gemm_tiles<8, 8>(s_x, 512, Wih, H_DIM, w * 128, lane, acc);
    // Phase 2: h @ Whh^T     (K = 256)
    gemm_tiles<8, 8>(s_h, 512, Whh, H_DIM, w * 128, lane, acc);

    // scatter C-tiles to gate LDS: lane half selects M rows 0-7 / 8-15
    const int half = (lane >> 4) & 1;
#pragma unroll
    for (int c = 0; c < 8; ++c) {
      int n = w * 128 + c * 16 + (lane & 15);
#pragma unroll
      for (int r = 0; r < 8; ++r) {
        int m = r + 8 * half;
        s_g[m * 1024 + n] = acc[c][r];
      }
    }
    __syncthreads();   // gates ready; also guards s_h rewrite below

    // fused LSTM cell: thread owns hidden column hc for all 16 rows
#pragma unroll
    for (int m = 0; m < 16; ++m) {
      float gi = s_g[m * 1024 + hc]        + bi;
      float gf = s_g[m * 1024 + 256 + hc]  + bf;
      float gg = s_g[m * 1024 + 512 + hc]  + bg;
      float go = s_g[m * 1024 + 768 + hc]  + bo;
      float cv = sigm(gf) * c_reg[m] + sigm(gi) * tanhf(gg);
      float hv = sigm(go) * tanhf(cv);
      c_reg[m] = cv;
      __bf16 hb = f2bf(hv);
      ((__bf16*)s_h)[m * H_DIM + hc] = hb;
      Hout[(size_t)(mBase + m) * T_STEPS * H_DIM + (size_t)t * H_DIM + hc] = hb;
    }
  }
}

// ---------------------------------------------------------------------------
// Kernel 3: FC (256->192 via WMMA) + mixture-density head.
// Block = 16 flattened (m,t) rows, 128 threads = 4 waves, 3 C-tiles each.
// ---------------------------------------------------------------------------
__global__ void fc_mdn_kernel(const __bf16* __restrict__ H1,
                              const __bf16* __restrict__ fcWb,
                              const float* __restrict__ fcb,
                              float* __restrict__ out) {
  extern __shared__ __align__(16) char smem[];
  char*  s_a = smem;                       // [16][512] bytes bf16
  float* s_y = (float*)(smem + 8192);      // [16][192] f32

  const int tid  = threadIdx.x;
  const int lane = tid & 31;
  const int w    = tid >> 5;               // 0..3
  const int rBase = blockIdx.x * 16;

  // stage 16x256 bf16 A tile
  for (int i = tid; i < 16 * 32; i += 128) {
    int m = i >> 5, co = i & 31;
    *(uint4*)(s_a + m * 512 + co * 16) =
        *(const uint4*)(H1 + (size_t)(rBase + m) * H_DIM + co * 8);
  }
  __syncthreads();

  const v8f vzero = {0.f, 0.f, 0.f, 0.f, 0.f, 0.f, 0.f, 0.f};
  v8f acc[3] = {vzero, vzero, vzero};
  gemm_tiles<8, 3>(s_a, 512, fcWb, H_DIM, w * 48, lane, acc);

  const int half = (lane >> 4) & 1;
#pragma unroll
  for (int c = 0; c < 3; ++c) {
    int n = w * 48 + c * 16 + (lane & 15);
#pragma unroll
    for (int r = 0; r < 8; ++r) {
      int m = r + 8 * half;
      s_y[m * OUT_DIM + n] = acc[c][r] + fcb[n];
    }
  }
  __syncthreads();

  // mixture-density head, one thread per row (deterministic hash RNG)
  if (tid < 16) {
    const int r = rBase + tid;
    const float* y = s_y + tid * OUT_DIM;

    float mx = y[0];
    for (int c = 1; c < NCOMP; ++c) mx = fmaxf(mx, y[c]);
    float se = 0.0f;
    for (int c = 0; c < NCOMP; ++c) se += __expf(y[c] - mx);
    float inv = 1.0f / se;

    unsigned h1 = hash_u32(0x9E3779B9u ^ (unsigned)r);
    unsigned h2 = hash_u32(h1 + 0x85EBCA6Bu);
    unsigned h3 = hash_u32(h2 + 0xC2B2AE35u);
    float u1 = (h1 >> 8) * (1.0f / 16777216.0f);
    float u2 = fmaxf((h2 >> 8) * (1.0f / 16777216.0f), 1e-7f);
    float u3 = (h3 >> 8) * (1.0f / 16777216.0f);

    // categorical sample from softmax(y[:64])
    int idx = NCOMP - 1;
    float cum = 0.0f;
    for (int c = 0; c < NCOMP; ++c) {
      cum += __expf(y[c] - mx) * inv;
      if (cum >= u1) { idx = c; break; }
    }
    float mu_s  = y[128 + idx];
    float sig_s = __expf(y[64 + idx]);
    float eps = sqrtf(-2.0f * __logf(u2)) * __cosf(6.28318530718f * u3);
    float xp  = mu_s + sig_s * eps;

    // p = sum_c gamma_c * exp(-0.5 d^2) * sig^-66 * (2pi)^-32
    const float INV_2PI_P32 = 2.8713e-26f;
    float p = 0.0f;
    for (int c = 0; c < NCOMP; ++c) {
      float gam = __expf(y[c] - mx) * inv;
      float d   = xp - y[128 + c];
      float k   = __expf(-0.5f * d * d - 66.0f * y[64 + c]) * INV_2PI_P32;
      p += gam * k;
    }
    out[r]        = xp;
    out[ROWS + r] = p;
  }
}

// ---------------------------------------------------------------------------
// Workspace layout (bytes)
// ---------------------------------------------------------------------------
#define WS_ZPAD 0UL                         // 131072*32 bf16  =  8,388,608
#define WS_WIH0 8388608UL                   // 1024*32  bf16   =     65,536
#define WS_WHH0 8454144UL                   // 1024*256 bf16   =    524,288
#define WS_WIH1 8978432UL                   // 1024*256 bf16   =    524,288
#define WS_WHH1 9502720UL                   // 1024*256 bf16   =    524,288
#define WS_FCW  10027008UL                  // 192*256  bf16   =     98,304
#define WS_H0   10125312UL                  // 131072*256 bf16 = 67,108,864
#define WS_H1   77234176UL                  // 131072*256 bf16 = 67,108,864

extern "C" void kernel_launch(void* const* d_in, const int* in_sizes, int n_in,
                              void* d_out, int out_size, void* d_ws, size_t ws_size,
                              hipStream_t stream) {
  (void)in_sizes; (void)n_in; (void)out_size; (void)ws_size;
  const float* z    = (const float*)d_in[0];
  const float* Wih0 = (const float*)d_in[1];
  const float* Whh0 = (const float*)d_in[2];
  const float* bih0 = (const float*)d_in[3];
  const float* bhh0 = (const float*)d_in[4];
  const float* Wih1 = (const float*)d_in[5];
  const float* Whh1 = (const float*)d_in[6];
  const float* bih1 = (const float*)d_in[7];
  const float* bhh1 = (const float*)d_in[8];
  const float* fcW  = (const float*)d_in[9];
  const float* fcb  = (const float*)d_in[10];

  char* ws = (char*)d_ws;
  __bf16* zpad  = (__bf16*)(ws + WS_ZPAD);
  __bf16* wih0p = (__bf16*)(ws + WS_WIH0);
  __bf16* whh0b = (__bf16*)(ws + WS_WHH0);
  __bf16* wih1b = (__bf16*)(ws + WS_WIH1);
  __bf16* whh1b = (__bf16*)(ws + WS_WHH1);
  __bf16* fcwb  = (__bf16*)(ws + WS_FCW);
  __bf16* h0    = (__bf16*)(ws + WS_H0);
  __bf16* h1    = (__bf16*)(ws + WS_H1);

  // 1) convert/pad weights + inputs to bf16
  prep_kernel<<<19776, 256, 0, stream>>>(z, Wih0, Whh0, Wih1, Whh1, fcW,
                                         zpad, wih0p, whh0b, wih1b, whh1b, fcwb);

  // 2) LSTM layers: 128 M-blocks of 16 sequences, 8 waves each, 80KB LDS
  const size_t lstm_smem = 16384 + 16 * 1024 * sizeof(float);  // 81920
  lstm_layer_kernel<<<M_SEQ / 16, 256, lstm_smem, stream>>>(
      zpad, 32, wih0p, whh0b, bih0, bhh0, h0);
  lstm_layer_kernel<<<M_SEQ / 16, 256, lstm_smem, stream>>>(
      h0, 256, wih1b, whh1b, bih1, bhh1, h1);

  // 3) FC + MDN head: 8192 row-tiles of 16, 4 waves, 20KB LDS
  const size_t fc_smem = 8192 + 16 * OUT_DIM * sizeof(float);  // 20480
  fc_mdn_kernel<<<ROWS / 16, 128, fc_smem, stream>>>(h1, fcwb, fcb,
                                                     (float*)d_out);
}
